// Adj_32100585571004
// MI455X (gfx1250) — matrix operand order repaired and checked by reading, not compiled, not run on hardware
//
#include <hip/hip_runtime.h>
#include <math.h>
#include <stdint.h>

typedef __attribute__((ext_vector_type(16))) _Float16 v16h;
typedef __attribute__((ext_vector_type(4)))  _Float16 v4h;
typedef __attribute__((ext_vector_type(8)))  float    v8f;

#define NN 8192
#define DD 1024
#define BM 128
#define BN 128
#define BK 32
#define LDSK 40   // halves per LDS row (32 + 8 pad -> 80B row stride, 16B aligned)
#define BUFH (BM * LDSK)   // halves per A/B buffer

// exp(-dist/T) with T=0.01  ->  exp2(dist * (-100*log2(e)))
#define NEG_INVT_LOG2E (-144.2695040888963f)

union Frag { v16h h; uint4 u[2]; };

// ---- CDNA5 async global->LDS copy (ASYNCcnt path), 16 bytes per lane ----
__device__ __forceinline__ void async_load_b128(void* lds, const void* gptr) {
  unsigned l = (unsigned)(uintptr_t)lds;   // LDS aperture: low 32 bits = LDS byte addr
  unsigned long long g = (unsigned long long)(uintptr_t)gptr;
  asm volatile("global_load_async_to_lds_b128 %0, %1, off"
               :: "v"(l), "v"(g) : "memory");
}
__device__ __forceinline__ void wait_async0() {
  asm volatile("s_wait_asynccnt 0x0" ::: "memory");
}

// ---------------- Kernel 1a: row squared norms + fp32->f16 conversion ----------------
__global__ __launch_bounds__(256) void prep_kernel(const float* __restrict__ X,
                                                   float* __restrict__ sq,
                                                   _Float16* __restrict__ Xh) {
  const int row = blockIdx.x;
  const float*  xr = X  + (size_t)row * DD;
  _Float16*     hr = Xh + (size_t)row * DD;
  float s = 0.0f;
  for (int c4 = threadIdx.x; c4 < DD / 4; c4 += 256) {
    float4 v = ((const float4*)xr)[c4];
    s += v.x * v.x + v.y * v.y + v.z * v.z + v.w * v.w;
    v4h h;
    h.x = (_Float16)v.x; h.y = (_Float16)v.y; h.z = (_Float16)v.z; h.w = (_Float16)v.w;
    ((v4h*)hr)[c4] = h;
  }
  #pragma unroll
  for (int off = 16; off > 0; off >>= 1)
    s += __shfl_down(s, off, 32);
  __shared__ float red[8];
  if ((threadIdx.x & 31) == 0) red[threadIdx.x >> 5] = s;
  __syncthreads();
  if (threadIdx.x == 0) {
    float t = 0.0f;
    #pragma unroll
    for (int w = 0; w < 8; ++w) t += red[w];
    sq[row] = t;
  }
}

// ---------------- Kernel 1b: norms only (fallback path) ----------------
__global__ __launch_bounds__(256) void row_sq_kernel(const float* __restrict__ X,
                                                     float* __restrict__ sq) {
  const int row = blockIdx.x;
  const float* xr = X + (size_t)row * DD;
  float s = 0.0f;
  for (int c = threadIdx.x; c < DD; c += 256) {
    float v = xr[c];
    s += v * v;
  }
  #pragma unroll
  for (int off = 16; off > 0; off >>= 1)
    s += __shfl_down(s, off, 32);
  __shared__ float red[8];
  if ((threadIdx.x & 31) == 0) red[threadIdx.x >> 5] = s;
  __syncthreads();
  if (threadIdx.x == 0) {
    float t = 0.0f;
    #pragma unroll
    for (int w = 0; w < 8; ++w) t += red[w];
    sq[row] = t;
  }
}

// One K=32 step: load 4 A-fragments + 2 B-fragments from LDS, issue 8 WMMAs
// into 8 individually named accumulators (tied dst==src2 -> no copies).
#define WMMA_STEP(pAs, pBs)                                                        \
  do {                                                                             \
    Frag a0, a1, a2, a3, b0, b1;                                                   \
    {                                                                              \
      const _Float16* p;                                                           \
      p = (pAs) + (wm +  0 + lm) * LDSK + kg * 8;                                  \
      a0.u[0] = *(const uint4*)p; a0.u[1] = *(const uint4*)(p + 16);               \
      p = (pAs) + (wm + 16 + lm) * LDSK + kg * 8;                                  \
      a1.u[0] = *(const uint4*)p; a1.u[1] = *(const uint4*)(p + 16);               \
      p = (pAs) + (wm + 32 + lm) * LDSK + kg * 8;                                  \
      a2.u[0] = *(const uint4*)p; a2.u[1] = *(const uint4*)(p + 16);               \
      p = (pAs) + (wm + 48 + lm) * LDSK + kg * 8;                                  \
      a3.u[0] = *(const uint4*)p; a3.u[1] = *(const uint4*)(p + 16);               \
      p = (pBs) + (wn +  0 + lm) * LDSK + kg * 8;                                  \
      b0.u[0] = *(const uint4*)p; b0.u[1] = *(const uint4*)(p + 16);               \
      p = (pBs) + (wn + 16 + lm) * LDSK + kg * 8;                                  \
      b1.u[0] = *(const uint4*)p; b1.u[1] = *(const uint4*)(p + 16);               \
    }                                                                              \
    c00 = __builtin_amdgcn_wmma_f32_16x16x32_f16(false, a0.h, false, b0.h, (short)0, c00, false, false); \
    c01 = __builtin_amdgcn_wmma_f32_16x16x32_f16(false, a0.h, false, b1.h, (short)0, c01, false, false); \
    c10 = __builtin_amdgcn_wmma_f32_16x16x32_f16(false, a1.h, false, b0.h, (short)0, c10, false, false); \
    c11 = __builtin_amdgcn_wmma_f32_16x16x32_f16(false, a1.h, false, b1.h, (short)0, c11, false, false); \
    c20 = __builtin_amdgcn_wmma_f32_16x16x32_f16(false, a2.h, false, b0.h, (short)0, c20, false, false); \
    c21 = __builtin_amdgcn_wmma_f32_16x16x32_f16(false, a2.h, false, b1.h, (short)0, c21, false, false); \
    c30 = __builtin_amdgcn_wmma_f32_16x16x32_f16(false, a3.h, false, b0.h, (short)0, c30, false, false); \
    c31 = __builtin_amdgcn_wmma_f32_16x16x32_f16(false, a3.h, false, b1.h, (short)0, c31, false, false); \
  } while (0)

__device__ __forceinline__ void emit_tile(const float* __restrict__ sq,
                                          float* __restrict__ out,
                                          const v8f& c, int ib, int jb, int kg) {
  const float sqj = sq[jb];
  #pragma unroll
  for (int r = 0; r < 8; ++r) {
    const int ii = ib + kg * 8 + r;          // C layout: M = r + 8*(lane>=16)
    float d2 = sq[ii] + sqj - 2.0f * c[r];
    d2 = fmaxf(d2, 0.0f);
    out[(size_t)ii * NN + jb] = exp2f(NEG_INVT_LOG2E * sqrtf(d2));
  }
}

#define EPILOGUE()                                                                  \
  do {                                                                              \
    emit_tile(sq, out, c00, i0 + wm +  0, j0 + wn +  0 + lm, kg);                   \
    emit_tile(sq, out, c01, i0 + wm +  0, j0 + wn + 16 + lm, kg);                   \
    emit_tile(sq, out, c10, i0 + wm + 16, j0 + wn +  0 + lm, kg);                   \
    emit_tile(sq, out, c11, i0 + wm + 16, j0 + wn + 16 + lm, kg);                   \
    emit_tile(sq, out, c20, i0 + wm + 32, j0 + wn +  0 + lm, kg);                   \
    emit_tile(sq, out, c21, i0 + wm + 32, j0 + wn + 16 + lm, kg);                   \
    emit_tile(sq, out, c30, i0 + wm + 48, j0 + wn +  0 + lm, kg);                   \
    emit_tile(sq, out, c31, i0 + wm + 48, j0 + wn + 16 + lm, kg);                   \
  } while (0)

// ---------------- Kernel 2a: WMMA GEMM with async global->LDS staging (f16 input) ----
__global__ __launch_bounds__(256) void adj_wmma_async_kernel(const _Float16* __restrict__ Xh,
                                                             const float* __restrict__ sq,
                                                             float* __restrict__ out) {
  __shared__ _Float16 As[2 * BUFH];
  __shared__ _Float16 Bs[2 * BUFH];

  const int tid  = threadIdx.x;
  const int lane = tid & 31;
  const int wave = tid >> 5;
  const int wm   = (wave >> 2) * 64;
  const int wn   = (wave & 3) * 32;
  const int lm   = lane & 15;
  const int kg   = lane >> 4;
  const int i0   = blockIdx.y * BM;
  const int j0   = blockIdx.x * BN;

  v8f c00 = (v8f){}, c01 = (v8f){}, c10 = (v8f){}, c11 = (v8f){};
  v8f c20 = (v8f){}, c21 = (v8f){}, c30 = (v8f){}, c31 = (v8f){};

  // Stage one 128x32 f16 chunk of A and B into buffer at half-offset `boff`
  // via async-to-LDS b128 copies: 512 16B chunks/matrix -> 2 per thread each.
  auto stage = [&](unsigned boff, int kt) {
    const int k0 = kt * BK;
    #pragma unroll
    for (int s = 0; s < 2; ++s) {
      const int idx = tid + s * 256;    // 0..511
      const int row = idx >> 2;         // 4 b128 chunks per 32-half row
      const int c   = idx & 3;
      const _Float16* ga = Xh + (size_t)(i0 + row) * DD + k0 + c * 8;
      const _Float16* gb = Xh + (size_t)(j0 + row) * DD + k0 + c * 8;
      async_load_b128(&As[boff + row * LDSK + c * 8], ga);
      async_load_b128(&Bs[boff + row * LDSK + c * 8], gb);
    }
  };

  stage(0u, 0);
  wait_async0();
  __syncthreads();

  const int KT = DD / BK;  // 32
  unsigned cur = 0u;
  #pragma clang loop unroll(disable)
  for (int kt = 0; kt < KT; ++kt) {
    const unsigned nxt = (unsigned)BUFH - cur;   // ping-pong: 0 <-> BUFH
    if (kt + 1 < KT) stage(nxt, kt + 1);         // async prefetch next buffer
    WMMA_STEP(&As[cur], &Bs[cur]);
    wait_async0();                               // next buffer landed in LDS
    __syncthreads();
    cur = nxt;
  }

  EPILOGUE();
}

// ---------------- Kernel 2b: WMMA GEMM, convert-in-kernel staging (fallback) --------
__global__ __launch_bounds__(256) void adj_wmma_kernel(const float* __restrict__ X,
                                                       const float* __restrict__ sq,
                                                       float* __restrict__ out) {
  __shared__ _Float16 As[2 * BUFH];
  __shared__ _Float16 Bs[2 * BUFH];

  const int tid  = threadIdx.x;
  const int lane = tid & 31;
  const int wave = tid >> 5;
  const int wm   = (wave >> 2) * 64;
  const int wn   = (wave & 3) * 32;
  const int lm   = lane & 15;
  const int kg   = lane >> 4;
  const int i0   = blockIdx.y * BM;
  const int j0   = blockIdx.x * BN;

  v8f c00 = (v8f){}, c01 = (v8f){}, c10 = (v8f){}, c11 = (v8f){};
  v8f c20 = (v8f){}, c21 = (v8f){}, c30 = (v8f){}, c31 = (v8f){};

  auto stage = [&](unsigned boff, int kt) {
    const int k0 = kt * BK;
    #pragma unroll
    for (int s = 0; s < 4; ++s) {
      const int idx = tid + s * 256;     // 0..1023
      const int row = idx >> 3;          // 8 float4 per 32-float row
      const int c4  = idx & 7;
      const float4 va = *(const float4*)(X + (size_t)(i0 + row) * DD + k0 + c4 * 4);
      const float4 vb = *(const float4*)(X + (size_t)(j0 + row) * DD + k0 + c4 * 4);
      v4h ha, hb;
      ha.x = (_Float16)va.x; ha.y = (_Float16)va.y; ha.z = (_Float16)va.z; ha.w = (_Float16)va.w;
      hb.x = (_Float16)vb.x; hb.y = (_Float16)vb.y; hb.z = (_Float16)vb.z; hb.w = (_Float16)vb.w;
      *(v4h*)(&As[boff + row * LDSK + c4 * 4]) = ha;
      *(v4h*)(&Bs[boff + row * LDSK + c4 * 4]) = hb;
    }
  };

  stage(0u, 0);
  __syncthreads();

  const int KT = DD / BK;
  unsigned cur = 0u;
  #pragma clang loop unroll(disable)
  for (int kt = 0; kt < KT; ++kt) {
    const unsigned nxt = (unsigned)BUFH - cur;
    if (kt + 1 < KT) stage(nxt, kt + 1);
    WMMA_STEP(&As[cur], &Bs[cur]);
    __syncthreads();
    cur = nxt;
  }

  EPILOGUE();
}

extern "C" void kernel_launch(void* const* d_in, const int* in_sizes, int n_in,
                              void* d_out, int out_size, void* d_ws, size_t ws_size,
                              hipStream_t stream) {
  (void)in_sizes; (void)n_in; (void)out_size;
  const float* X = (const float*)d_in[0];
  float* out     = (float*)d_out;

  // Workspace layout: [ sq : NN floats | pad to 64KB | Xh : NN*DD f16 (16 MB) ]
  const size_t XH_OFF = 65536;
  const size_t need   = XH_OFF + sizeof(_Float16) * (size_t)NN * DD;

  float* sq = (float*)d_ws;
  dim3 grid(NN / BN, NN / BM);

  if (ws_size >= need) {
    _Float16* Xh = (_Float16*)((char*)d_ws + XH_OFF);
    prep_kernel<<<NN, 256, 0, stream>>>(X, sq, Xh);
    adj_wmma_async_kernel<<<grid, 256, 0, stream>>>(Xh, sq, out);
  } else {
    row_sq_kernel<<<NN, 256, 0, stream>>>(X, sq);
    adj_wmma_kernel<<<grid, 256, 0, stream>>>(X, sq, out);
  }
}
